// TensorProductConvBlock_55405078118473
// MI455X (gfx1250) — matrix operand order repaired and checked by reading, not compiled, not run on hardware
//
#include <hip/hip_runtime.h>
#include <math.h>

// ---------------------------------------------------------------------------
// Types for CDNA5 WMMA (wave32, 16x16x32 f16 -> f32)
// ---------------------------------------------------------------------------
typedef _Float16 half_t;
typedef __attribute__((ext_vector_type(16))) _Float16 v16h;
typedef __attribute__((ext_vector_type(8)))  _Float16 v8h;
typedef __attribute__((ext_vector_type(8)))  float    v8f;

#define WAVE_SYNC() asm volatile("s_wait_dscnt 0x0" ::: "memory")

#define WPB 8            /* waves per block (256 threads) */
#define LDSPAD 8         /* half-elements of row padding for LDS weight tiles */

// A-matrix fragment (16x32 f16). Per ISA layout: lanes 0-15 hold row=lane,
// K = {0..7, 16..23}; lanes 16-31 hold row=lane-16, K = {8..15, 24..31}.
__device__ __forceinline__ v16h a_frag(const half_t* __restrict__ row, int ks, int lane) {
    const half_t* p = row + ks * 32 + ((lane & 16) ? 8 : 0);
    v8h lo = *(const v8h*)(p);
    v8h hi = *(const v8h*)(p + 16);
    return __builtin_shufflevector(lo, hi, 0,1,2,3,4,5,6,7,8,9,10,11,12,13,14,15);
}

// B-matrix fragment (32x16 f16) from row-major W[64][ldn]: lane holds column
// nt*16 + lane%16, K = {0..7, 16..23} (lanes 0-15) / {8..15, 24..31} (lanes 16-31).
__device__ __forceinline__ v16h b_frag(const half_t* __restrict__ W, int ldn, int ks, int nt, int lane) {
    const half_t* p = W + (size_t)(ks * 32 + ((lane & 16) ? 8 : 0)) * ldn + nt * 16 + (lane & 15);
    v16h b;
    #pragma unroll
    for (int s = 0; s < 8; ++s) {
        b[s] = p[(size_t)s * ldn];
        b[8 + s] = p[(size_t)(16 + s) * ldn];
    }
    return b;
}

__device__ __forceinline__ v8f wmma16(v16h a, v16h b, v8f c) {
    return __builtin_amdgcn_wmma_f32_16x16x32_f16(false, a, false, b, (short)0, c, false, false);
}

// ---------------------------------------------------------------------------
// Constants
// ---------------------------------------------------------------------------
__device__ __forceinline__ float sq3()  { return 1.7320508075688772f; }
#define RBF_COEFF (-0.6805555555555556f)   /* -0.5 / (6/7)^2 */
#define PW0 0.22360679774997896f           /* 1/sqrt(20) */
#define PW1 0.3535533905932738f            /* sqrt(0.125) */
#define P1E 0.8660254037844386f            /* sqrt(0.75) */
#define INV3 0.5773502691896258f           /* 1/sqrt(3) */
#define C_A 0.31622776601683794f           /* 1/sqrt(10) */
#define C_B 0.3651483716701107f            /* 1/sqrt(7.5) */
#define INV6 0.4082482904638631f           /* 1/sqrt(6) */

// ---------------------------------------------------------------------------
// Small utility kernels
// ---------------------------------------------------------------------------
__global__ void zero_kernel(float* __restrict__ p, int count) {
    int i = blockIdx.x * blockDim.x + threadIdx.x;
    if (i < count) p[i] = 0.0f;
}

// Convert f32 weight [K][N] to f16 [64][N] with zero K-padding.
__global__ void conv_half_kernel(const float* __restrict__ W, half_t* __restrict__ Wh, int K, int N) {
    int i = blockIdx.x * blockDim.x + threadIdx.x;
    if (i < 64 * N) {
        int k = i / N, j = i % N;
        Wh[i] = (k < K) ? (half_t)W[k * N + j] : (half_t)0.0f;
    }
}

__global__ void count_kernel(const int* __restrict__ nbr, int P, float* __restrict__ cnt) {
    int e = blockIdx.x * blockDim.x + threadIdx.x;
    if (e < 2 * P) {
        int src = (e < P) ? nbr[2 * e] : nbr[2 * (e - P) + 1];
        atomicAdd(&cnt[src], 1.0f);
    }
}

// node = tanh(z @ w1 + b1) @ w2 + b2   (32 -> 16 -> 16)
__global__ void node_mlp_kernel(const float* __restrict__ z,
                                const float* __restrict__ w1, const float* __restrict__ b1,
                                const float* __restrict__ w2, const float* __restrict__ b2,
                                float* __restrict__ node0, int n) {
    int i = blockIdx.x * blockDim.x + threadIdx.x;
    if (i >= n) return;
    float t[16];
    #pragma unroll 4
    for (int j = 0; j < 16; ++j) {
        float a = b1[j];
        for (int k = 0; k < 32; ++k) a += z[i * 32 + k] * w1[k * 16 + j];
        t[j] = tanhf(a);
    }
    #pragma unroll 4
    for (int j = 0; j < 16; ++j) {
        float a = b2[j];
        for (int k = 0; k < 16; ++k) a += t[k] * w2[k * 16 + j];
        node0[i * 16 + j] = a;
    }
}

__global__ void finalize28_kernel(const float* __restrict__ node0, const float* __restrict__ acc,
                                  const float* __restrict__ cnt, float* __restrict__ node1, int n) {
    int idx = blockIdx.x * blockDim.x + threadIdx.x;
    if (idx < n * 28) {
        int i = idx / 28, f = idx % 28;
        float base = (f < 16) ? node0[i * 16 + f] : 0.0f;
        node1[idx] = base + acc[idx] / fmaxf(cnt[i], 1.0f);
    }
}

__global__ void finalize40_kernel(const float* __restrict__ node1, const float* __restrict__ acc,
                                  const float* __restrict__ cnt, float* __restrict__ out, int n) {
    int idx = blockIdx.x * blockDim.x + threadIdx.x;
    if (idx < n * 40) {
        int i = idx / 40, f = idx % 40;
        float base = (f < 28) ? node1[i * 28 + f] : 0.0f;
        out[idx] = base + acc[idx] / fmaxf(cnt[i], 1.0f);
    }
}

// ---------------------------------------------------------------------------
// Layer 0: fused edge features + MLP (48->48->320 via WMMA) + tensor product
// + segment scatter. One wave handles a tile of 16 edges; weights staged in
// LDS once per block and shared by all 8 waves.
// ---------------------------------------------------------------------------
__global__ __launch_bounds__(256)
void layer0_kernel(const int* __restrict__ nbr, const float* __restrict__ xyz,
                   const float* __restrict__ node0,
                   const float* __restrict__ ew1, const float* __restrict__ eb1,
                   const float* __restrict__ ew2, const float* __restrict__ eb2,
                   const half_t* __restrict__ w1h, const float* __restrict__ fb1,
                   const half_t* __restrict__ w2h, const float* __restrict__ fb2,
                   float* __restrict__ acc0, int P) {
    const int E = 2 * P;
    const int lane = threadIdx.x & 31;
    const int wid = threadIdx.x >> 5;
    const int cl = lane & 15;
    const int rowoff = (lane & 16) ? 8 : 0;
    const int w4 = cl & 3;

    __shared__ __align__(16) half_t s_ea [WPB][16][64];
    __shared__ __align__(16) half_t s_hid[WPB][16][64];
    __shared__ float s_tp [WPB][16][28];
    __shared__ float s_sh1[WPB][16][4];
    __shared__ __align__(16) half_t s_w1[64 * (48 + LDSPAD)];
    __shared__ __align__(16) half_t s_w2[64 * (320 + LDSPAD)];

    // ---- stage fc weights into LDS (block-cooperative, once) ----
    for (int idx = threadIdx.x; idx < 64 * 48; idx += blockDim.x)
        s_w1[(idx / 48) * (48 + LDSPAD) + (idx % 48)] = w1h[idx];
    for (int idx = threadIdx.x; idx < 64 * 320; idx += blockDim.x)
        s_w2[(idx / 320) * (320 + LDSPAD) + (idx % 320)] = w2h[idx];
    __syncthreads();

    half_t (*ea)[64]  = s_ea[wid];
    half_t (*hid)[64] = s_hid[wid];
    float (*tp)[28]   = s_tp[wid];
    float (*sh1l)[4]  = s_sh1[wid];

    const int ntiles = (E + 15) / 16;
    for (int tile = blockIdx.x * WPB + wid; tile < ntiles; tile += gridDim.x * WPB) {
        // ---- zero tp + K-pad columns, build edge rows ----
        for (int idx = lane; idx < 16 * 28; idx += 32) tp[idx / 28][idx % 28] = 0.0f;
        for (int idx = lane; idx < 16 * 16; idx += 32) {
            ea [idx >> 4][48 + (idx & 15)] = (half_t)0.0f;
            hid[idx >> 4][48 + (idx & 15)] = (half_t)0.0f;
        }
        int e0 = tile * 16 + lane;
        if (lane < 16 && e0 < E) {
            int src, dst;
            if (e0 < P) { src = nbr[2 * e0];           dst = nbr[2 * e0 + 1]; }
            else        { src = nbr[2 * (e0 - P) + 1]; dst = nbr[2 * (e0 - P)]; }
            float rx = xyz[3 * dst + 0] - xyz[3 * src + 0];
            float ry = xyz[3 * dst + 1] - xyz[3 * src + 1];
            float rz = xyz[3 * dst + 2] - xyz[3 * src + 2];
            float d = sqrtf(rx * rx + ry * ry + rz * rz);
            float inv = 1.0f / fmaxf(d, 1e-9f);
            sh1l[lane][0] = sq3() * rx * inv;
            sh1l[lane][1] = sq3() * ry * inv;
            sh1l[lane][2] = sq3() * rz * inv;
            float rb[8];
            #pragma unroll
            for (int j = 0; j < 8; ++j) {
                float t = d - (6.0f / 7.0f) * (float)j;
                rb[j] = expf(RBF_COEFF * t * t);
            }
            float eh[16];
            #pragma unroll 4
            for (int j = 0; j < 16; ++j) {
                float a = eb1[j];
                for (int k = 0; k < 8; ++k) a += rb[k] * ew1[k * 16 + j];
                eh[j] = fmaxf(a, 0.0f);
            }
            #pragma unroll 4
            for (int j = 0; j < 16; ++j) {
                float a = eb2[j];
                for (int k = 0; k < 16; ++k) a += eh[k] * ew2[k * 16 + j];
                ea[lane][j] = (half_t)a;
            }
            #pragma unroll 4
            for (int j = 0; j < 16; ++j) {
                ea[lane][16 + j] = (half_t)node0[src * 16 + j];
                ea[lane][32 + j] = (half_t)node0[dst * 16 + j];
            }
        }
        WAVE_SYNC();

        // ---- GEMM1: hid = relu(ea @ w1 + fb1), N = 48 ----
        {
            v16h a0 = a_frag(ea[cl], 0, lane);     // hoisted: invariant across nt
            v16h a1 = a_frag(ea[cl], 1, lane);
            #pragma unroll
            for (int nt = 0; nt < 3; ++nt) {
                v8f c = {};
                c = wmma16(a0, b_frag(s_w1, 48 + LDSPAD, 0, nt, lane), c);
                c = wmma16(a1, b_frag(s_w1, 48 + LDSPAD, 1, nt, lane), c);
                int gcol = nt * 16 + cl;
                float bb = fb1[gcol];
                #pragma unroll
                for (int v = 0; v < 8; ++v)
                    hid[v + rowoff][gcol] = (half_t)fmaxf(c[v] + bb, 0.0f);
            }
        }
        WAVE_SYNC();

        // ---- GEMM2 (N = 320) fused with tensor-product contraction ----
        float o0acc[8], sbacc[8];
        #pragma unroll
        for (int v = 0; v < 8; ++v) { o0acc[v] = 0.0f; sbacc[v] = 0.0f; }

        {
            v16h a0 = a_frag(hid[cl], 0, lane);    // hoisted: invariant across nt
            v16h a1 = a_frag(hid[cl], 1, lane);
            for (int nt = 0; nt < 20; ++nt) {
                v8f c = {};
                c = wmma16(a0, b_frag(s_w2, 320 + LDSPAD, 0, nt, lane), c);
                c = wmma16(a1, b_frag(s_w2, 320 + LDSPAD, 1, nt, lane), c);
                float bb = fb2[nt * 16 + cl];
                if (nt < 16) {              // Wa[u=nt][col]: out0e
                    int u = nt;
                    #pragma unroll
                    for (int v = 0; v < 8; ++v) {
                        float x = (float)ea[v + rowoff][32 + u];
                        o0acc[v] += x * (c[v] + bb);
                    }
                } else {                    // Wb: out1o, u = 4*(nt-16) + cl/4
                    int u = 4 * (nt - 16) + (cl >> 2);
                    #pragma unroll
                    for (int v = 0; v < 8; ++v) {
                        float x = (float)ea[v + rowoff][32 + u];
                        sbacc[v] += x * (c[v] + bb);
                    }
                }
            }
        }
        #pragma unroll
        for (int v = 0; v < 8; ++v) {
            int m = v + rowoff;
            tp[m][cl] = 0.25f * o0acc[v];                   // unique (edge,col) owner
            float s0 = 0.25f * sbacc[v];
            #pragma unroll
            for (int mx = 0; mx < 3; ++mx)                  // lanes share (edge,w4): LDS atomics
                atomicAdd(&tp[m][16 + w4 * 3 + mx], s0 * sh1l[m][mx]);
        }
        WAVE_SYNC();

        // ---- flush tile to global segment accumulators ----
        for (int idx = lane; idx < 16 * 28; idx += 32) {
            int ed = idx / 28, f = idx % 28;
            int e = tile * 16 + ed;
            if (e < E) {
                int src = (e < P) ? nbr[2 * e] : nbr[2 * (e - P) + 1];
                atomicAdd(&acc0[src * 28 + f], tp[ed][f]);
            }
        }
        WAVE_SYNC();
    }
}

// ---------------------------------------------------------------------------
// Layer 1: 48->48->432 via WMMA + full equivariant contraction.
// ---------------------------------------------------------------------------
__global__ __launch_bounds__(256)
void layer1_kernel(const int* __restrict__ nbr, const float* __restrict__ xyz,
                   const float* __restrict__ node1,
                   const float* __restrict__ ew1, const float* __restrict__ eb1,
                   const float* __restrict__ ew2, const float* __restrict__ eb2,
                   const half_t* __restrict__ w1h, const float* __restrict__ fb1,
                   const half_t* __restrict__ w2h, const float* __restrict__ fb2,
                   float* __restrict__ acc1, int P) {
    const int E = 2 * P;
    const int lane = threadIdx.x & 31;
    const int wid = threadIdx.x >> 5;
    const int cl = lane & 15;
    const int rowoff = (lane & 16) ? 8 : 0;
    const int w4 = cl & 3;

    __shared__ __align__(16) half_t s_ea [WPB][16][64];
    __shared__ __align__(16) half_t s_hid[WPB][16][64];
    __shared__ float s_tp [WPB][16][40];
    // per-edge geometry scratch: [0..2]=sh1, [3..14]=v(u*3+i), [15..18]=vs1,
    // [19..30]=cross(u*3+k)/sqrt6, [31..42]=t112(u*3+j)
    __shared__ float s_geo[WPB][16][44];
    __shared__ __align__(16) half_t s_w1[64 * (48 + LDSPAD)];
    __shared__ __align__(16) half_t s_w2[64 * (432 + LDSPAD)];

    for (int idx = threadIdx.x; idx < 64 * 48; idx += blockDim.x)
        s_w1[(idx / 48) * (48 + LDSPAD) + (idx % 48)] = w1h[idx];
    for (int idx = threadIdx.x; idx < 64 * 432; idx += blockDim.x)
        s_w2[(idx / 432) * (432 + LDSPAD) + (idx % 432)] = w2h[idx];
    __syncthreads();

    half_t (*ea)[64]  = s_ea[wid];
    half_t (*hid)[64] = s_hid[wid];
    float (*tp)[40]   = s_tp[wid];
    float (*geo)[44]  = s_geo[wid];

    const int ntiles = (E + 15) / 16;
    for (int tile = blockIdx.x * WPB + wid; tile < ntiles; tile += gridDim.x * WPB) {
        for (int idx = lane; idx < 16 * 40; idx += 32) tp[idx / 40][idx % 40] = 0.0f;
        for (int idx = lane; idx < 16 * 16; idx += 32) {
            ea [idx >> 4][48 + (idx & 15)] = (half_t)0.0f;
            hid[idx >> 4][48 + (idx & 15)] = (half_t)0.0f;
        }
        int e0 = tile * 16 + lane;
        if (lane < 16 && e0 < E) {
            int src, dst;
            if (e0 < P) { src = nbr[2 * e0];           dst = nbr[2 * e0 + 1]; }
            else        { src = nbr[2 * (e0 - P) + 1]; dst = nbr[2 * (e0 - P)]; }
            float rx = xyz[3 * dst + 0] - xyz[3 * src + 0];
            float ry = xyz[3 * dst + 1] - xyz[3 * src + 1];
            float rz = xyz[3 * dst + 2] - xyz[3 * src + 2];
            float d = sqrtf(rx * rx + ry * ry + rz * rz);
            float inv = 1.0f / fmaxf(d, 1e-9f);
            float hx = rx * inv, hy = ry * inv, hz = rz * inv;
            float s1x = sq3() * hx, s1y = sq3() * hy, s1z = sq3() * hz;
            geo[lane][0] = s1x; geo[lane][1] = s1y; geo[lane][2] = s1z;
            const float s15 = 3.872983346207417f, s5 = 2.23606797749979f;
            float sh2[5] = { s15 * hx * hz, s15 * hx * hy,
                             s5 * (hy * hy - 0.5f * (hx * hx + hz * hz)),
                             s15 * hy * hz, 0.5f * s15 * (hz * hz - hx * hx) };
            float rb[8];
            #pragma unroll
            for (int j = 0; j < 8; ++j) {
                float t = d - (6.0f / 7.0f) * (float)j;
                rb[j] = expf(RBF_COEFF * t * t);
            }
            float eh[16];
            #pragma unroll 4
            for (int j = 0; j < 16; ++j) {
                float a = eb1[j];
                for (int k = 0; k < 8; ++k) a += rb[k] * ew1[k * 16 + j];
                eh[j] = fmaxf(a, 0.0f);
            }
            #pragma unroll 4
            for (int j = 0; j < 16; ++j) {
                float a = eb2[j];
                for (int k = 0; k < 16; ++k) a += eh[k] * ew2[k * 16 + j];
                ea[lane][j] = (half_t)a;
            }
            #pragma unroll 4
            for (int j = 0; j < 16; ++j) {
                ea[lane][16 + j] = (half_t)node1[src * 28 + j];
                ea[lane][32 + j] = (half_t)node1[dst * 28 + j];
            }
            #pragma unroll
            for (int u = 0; u < 4; ++u) {
                float v0 = node1[dst * 28 + 16 + u * 3 + 0];
                float v1 = node1[dst * 28 + 16 + u * 3 + 1];
                float v2 = node1[dst * 28 + 16 + u * 3 + 2];
                geo[lane][3 + u * 3 + 0] = v0;
                geo[lane][3 + u * 3 + 1] = v1;
                geo[lane][3 + u * 3 + 2] = v2;
                geo[lane][15 + u] = v0 * s1x + v1 * s1y + v2 * s1z;   // vs1
                geo[lane][19 + u * 3 + 0] = (v1 * s1z - v2 * s1y) * INV6;  // cross/sqrt6
                geo[lane][19 + u * 3 + 1] = (v2 * s1x - v0 * s1z) * INV6;
                geo[lane][19 + u * 3 + 2] = (v0 * s1y - v1 * s1x) * INV6;
                // t112[u][j] = sum_{i,k} v_i * sh2_k * C112[i][j][k]
                geo[lane][31 + u * 3 + 0] = C_A * (v2 * sh2[0] + v1 * sh2[1])
                                            - 0.5f * C_B * v0 * sh2[2] - C_A * v0 * sh2[4];
                geo[lane][31 + u * 3 + 1] = C_A * v0 * sh2[1] + C_B * v1 * sh2[2] + C_A * v2 * sh2[3];
                geo[lane][31 + u * 3 + 2] = C_A * v0 * sh2[0] - 0.5f * C_B * v2 * sh2[2]
                                            + C_A * v1 * sh2[3] + C_A * v2 * sh2[4];
            }
        }
        WAVE_SYNC();

        {
            v16h a0 = a_frag(ea[cl], 0, lane);
            v16h a1 = a_frag(ea[cl], 1, lane);
            #pragma unroll
            for (int nt = 0; nt < 3; ++nt) {
                v8f c = {};
                c = wmma16(a0, b_frag(s_w1, 48 + LDSPAD, 0, nt, lane), c);
                c = wmma16(a1, b_frag(s_w1, 48 + LDSPAD, 1, nt, lane), c);
                int gcol = nt * 16 + cl;
                float bb = fb1[gcol];
                #pragma unroll
                for (int v = 0; v < 8; ++v)
                    hid[v + rowoff][gcol] = (half_t)fmaxf(c[v] + bb, 0.0f);
            }
        }
        WAVE_SYNC();

        float o0acc[8], sbacc[8];
        #pragma unroll
        for (int v = 0; v < 8; ++v) { o0acc[v] = 0.0f; sbacc[v] = 0.0f; }

        {
            v16h a0 = a_frag(hid[cl], 0, lane);
            v16h a1 = a_frag(hid[cl], 1, lane);
            for (int nt = 0; nt < 27; ++nt) {
                v8f c = {};
                c = wmma16(a0, b_frag(s_w2, 432 + LDSPAD, 0, nt, lane), c);
                c = wmma16(a1, b_frag(s_w2, 432 + LDSPAD, 1, nt, lane), c);
                float bb = fb2[nt * 16 + cl];
                if (nt < 16) {                       // Wa
                    int u = nt;
                    #pragma unroll
                    for (int v = 0; v < 8; ++v) {
                        float s = (float)ea[v + rowoff][32 + u];
                        o0acc[v] += s * (c[v] + bb);
                    }
                } else if (nt < 20) {                // Wb
                    int u = 4 * (nt - 16) + (cl >> 2);
                    #pragma unroll
                    for (int v = 0; v < 8; ++v) {
                        float s = (float)ea[v + rowoff][32 + u];
                        sbacc[v] += s * (c[v] + bb);
                    }
                } else if (nt == 20) {               // Wc: out1o += pw1/sqrt3 * v[u][m]*Wc
                    int u = cl >> 2;
                    #pragma unroll
                    for (int v = 0; v < 8; ++v) {
                        int m = v + rowoff;
                        float cw = c[v] + bb;
                        #pragma unroll
                        for (int mx = 0; mx < 3; ++mx)
                            atomicAdd(&tp[m][16 + w4 * 3 + mx], PW1 * INV3 * geo[m][3 + u * 3 + mx] * cw);
                    }
                } else if (nt < 25) {                // Wd: out0e += (1/sqrt3)*vs1[u]*Wd
                    int u = nt - 21;
                    #pragma unroll
                    for (int v = 0; v < 8; ++v) {
                        float cw = c[v] + bb;
                        o0acc[v] += INV3 * geo[v + rowoff][15 + u] * cw;
                    }
                } else if (nt == 25) {               // We: out1e += sqrt(.75)*cross6[u][k]*We
                    int u = cl >> 2;
                    #pragma unroll
                    for (int v = 0; v < 8; ++v) {
                        int m = v + rowoff;
                        float cw = c[v] + bb;
                        #pragma unroll
                        for (int mx = 0; mx < 3; ++mx)
                            atomicAdd(&tp[m][28 + w4 * 3 + mx], P1E * geo[m][19 + u * 3 + mx] * cw);
                    }
                } else {                             // Wf: out1o += pw1*t112[u][j]*Wf
                    int u = cl >> 2;
                    #pragma unroll
                    for (int v = 0; v < 8; ++v) {
                        int m = v + rowoff;
                        float cw = c[v] + bb;
                        #pragma unroll
                        for (int mx = 0; mx < 3; ++mx)
                            atomicAdd(&tp[m][16 + w4 * 3 + mx], PW1 * geo[m][31 + u * 3 + mx] * cw);
                    }
                }
            }
        }
        #pragma unroll
        for (int v = 0; v < 8; ++v) {
            int m = v + rowoff;
            tp[m][cl] = PW0 * o0acc[v];                    // out0e (unique owner)
            float s0 = PW1 * INV3 * sbacc[v];              // Wb partials
            #pragma unroll
            for (int mx = 0; mx < 3; ++mx)
                atomicAdd(&tp[m][16 + w4 * 3 + mx], s0 * geo[m][mx]);
        }
        WAVE_SYNC();

        for (int idx = lane; idx < 16 * 40; idx += 32) {
            int ed = idx / 40, f = idx % 40;
            int e = tile * 16 + ed;
            if (e < E) {
                int src = (e < P) ? nbr[2 * e] : nbr[2 * (e - P) + 1];
                atomicAdd(&acc1[src * 40 + f], tp[ed][f]);
            }
        }
        WAVE_SYNC();
    }
}

// ---------------------------------------------------------------------------
// Host-side launch
// ---------------------------------------------------------------------------
extern "C" void kernel_launch(void* const* d_in, const int* in_sizes, int n_in,
                              void* d_out, int out_size, void* d_ws, size_t ws_size,
                              hipStream_t stream) {
    const float* cg_z   = (const float*)d_in[0];
    const float* cg_xyz = (const float*)d_in[1];
    const int*   nbr    = (const int*)d_in[2];
    const float* node_w1 = (const float*)d_in[3];
    const float* node_b1 = (const float*)d_in[4];
    const float* node_w2 = (const float*)d_in[5];
    const float* node_b2 = (const float*)d_in[6];
    const float* edge_w1 = (const float*)d_in[7];
    const float* edge_b1 = (const float*)d_in[8];
    const float* edge_w2 = (const float*)d_in[9];
    const float* edge_b2 = (const float*)d_in[10];
    const float* fc0_w1 = (const float*)d_in[11];
    const float* fc0_b1 = (const float*)d_in[12];
    const float* fc0_w2 = (const float*)d_in[13];
    const float* fc0_b2 = (const float*)d_in[14];
    const float* fc1_w1 = (const float*)d_in[15];
    const float* fc1_b1 = (const float*)d_in[16];
    const float* fc1_w2 = (const float*)d_in[17];
    const float* fc1_b2 = (const float*)d_in[18];

    const int n = in_sizes[0] / 32;       // nodes
    const int P = in_sizes[2] / 2;        // pairs; E = 2P
    const int E = 2 * P;

    // workspace carve-up (aligned to 256B)
    char* w = (char*)d_ws;
    size_t off = 0;
    auto carve = [&](size_t bytes) -> void* {
        void* p = w + off;
        off += (bytes + 255) & ~(size_t)255;
        return p;
    };
    float* node0 = (float*)carve((size_t)n * 16 * 4);
    float* node1 = (float*)carve((size_t)n * 28 * 4);
    float* zblk  = (float*)carve((size_t)n * (28 + 40 + 1) * 4); // acc0|acc1|cnt contiguous
    float* acc0 = zblk;
    float* acc1 = zblk + (size_t)n * 28;
    float* cnt  = zblk + (size_t)n * 68;
    half_t* w1h0 = (half_t*)carve((size_t)64 * 48 * 2);
    half_t* w2h0 = (half_t*)carve((size_t)64 * 320 * 2);
    half_t* w1h1 = (half_t*)carve((size_t)64 * 48 * 2);
    half_t* w2h1 = (half_t*)carve((size_t)64 * 432 * 2);

    const int T = 256;
    int zc = n * 69;
    zero_kernel<<<(zc + T - 1) / T, T, 0, stream>>>(zblk, zc);
    conv_half_kernel<<<(64 * 48  + T - 1) / T, T, 0, stream>>>(fc0_w1, w1h0, 48, 48);
    conv_half_kernel<<<(64 * 320 + T - 1) / T, T, 0, stream>>>(fc0_w2, w2h0, 48, 320);
    conv_half_kernel<<<(64 * 48  + T - 1) / T, T, 0, stream>>>(fc1_w1, w1h1, 48, 48);
    conv_half_kernel<<<(64 * 432 + T - 1) / T, T, 0, stream>>>(fc1_w2, w2h1, 48, 432);
    count_kernel<<<(E + T - 1) / T, T, 0, stream>>>(nbr, P, cnt);
    node_mlp_kernel<<<(n + T - 1) / T, T, 0, stream>>>(cg_z, node_w1, node_b1, node_w2, node_b2, node0, n);

    layer0_kernel<<<1024, 256, 0, stream>>>(nbr, cg_xyz, node0,
                                            edge_w1, edge_b1, edge_w2, edge_b2,
                                            w1h0, fc0_b1, w2h0, fc0_b2, acc0, P);
    finalize28_kernel<<<(n * 28 + T - 1) / T, T, 0, stream>>>(node0, acc0, cnt, node1, n);

    layer1_kernel<<<1024, 256, 0, stream>>>(nbr, cg_xyz, node1,
                                            edge_w1, edge_b1, edge_w2, edge_b2,
                                            w1h1, fc1_b1, w2h1, fc1_b2, acc1, P);
    finalize40_kernel<<<(n * 40 + T - 1) / T, T, 0, stream>>>(node1, acc1, cnt, (float*)d_out, n);
}
